// CrossLayerTranscoder_46686294507772
// MI455X (gfx1250) — compile-verified, matrix operand order checked by reading, not run on hardware
//
#include <hip/hip_runtime.h>

// Problem constants (from reference)
#define ROWS   8192      // B*S
#define DIN    2048
#define HID    16384
#define DOUT   2048
#define TOPK   64

// Encoder tiling
#define M_TILE   64
#define H_CHUNK  256
#define CHUNKS   (HID / H_CHUNK)   // 64
#define NCAND    (CHUNKS * TOPK)   // 4096 candidates per row
#define A_STRIDE 2056              // DIN + 8 bf16 pad: 4112B row stride = 1028 DWORDs
                                   // == 4 (mod 64 banks) -> conflict-free ds_load_b128

typedef __attribute__((ext_vector_type(16))) __bf16 v16bf;
typedef __attribute__((ext_vector_type(8)))  __bf16 bf16x8;
typedef __attribute__((ext_vector_type(4)))  __bf16 bf16x4;
typedef __attribute__((ext_vector_type(8)))  float  v8f;
typedef __attribute__((ext_vector_type(4)))  float  f4;

static __device__ __forceinline__ v16bf cat8(bf16x8 lo, bf16x8 hi) {
  return __builtin_shufflevector(lo, hi, 0,1,2,3,4,5,6,7,8,9,10,11,12,13,14,15);
}
static __device__ __forceinline__ float wred_max(float v) {
  #pragma unroll
  for (int m = 16; m > 0; m >>= 1) v = fmaxf(v, __shfl_xor(v, m, 32));
  return v;
}
static __device__ __forceinline__ int wred_add(int v) {
  #pragma unroll
  for (int m = 16; m > 0; m >>= 1) v += __shfl_xor(v, m, 32);
  return v;
}
static __device__ __forceinline__ unsigned short f2bf_bits(float f) {
  __bf16 h = (__bf16)f;
  return __builtin_bit_cast(unsigned short, h);
}
static __device__ __forceinline__ float bf_bits2f(unsigned short u) {
  __bf16 h = __builtin_bit_cast(__bf16, u);
  return (float)h;
}

// Fragment bundle for one K step (4 A-groups from LDS, 2 B-tiles from global)
struct Frags {
  v16bf a0, a1, a2, a3, b0, b1;
};

static __device__ __forceinline__ Frags load_frags(
    const __bf16* ag0, const __bf16* ag1, const __bf16* ag2, const __bf16* ag3,
    const __bf16* w0, const __bf16* w1, int kk) {
  Frags f;
  f.b0 = cat8(*(const bf16x8*)(w0 + kk),  *(const bf16x8*)(w0 + kk + 16));
  f.b1 = cat8(*(const bf16x8*)(w1 + kk),  *(const bf16x8*)(w1 + kk + 16));
  f.a0 = cat8(*(const bf16x8*)(ag0 + kk), *(const bf16x8*)(ag0 + kk + 16));
  f.a1 = cat8(*(const bf16x8*)(ag1 + kk), *(const bf16x8*)(ag1 + kk + 16));
  f.a2 = cat8(*(const bf16x8*)(ag2 + kk), *(const bf16x8*)(ag2 + kk + 16));
  f.a3 = cat8(*(const bf16x8*)(ag3 + kk), *(const bf16x8*)(ag3 + kk + 16));
  return f;
}

static __device__ __forceinline__ void do_wmma(const Frags& f, v8f acc[4][2]) {
  acc[0][0] = __builtin_amdgcn_wmma_f32_16x16x32_bf16(
      false, f.a0, false, f.b0, (short)0, acc[0][0], false, false);
  acc[0][1] = __builtin_amdgcn_wmma_f32_16x16x32_bf16(
      false, f.a0, false, f.b1, (short)0, acc[0][1], false, false);
  acc[1][0] = __builtin_amdgcn_wmma_f32_16x16x32_bf16(
      false, f.a1, false, f.b0, (short)0, acc[1][0], false, false);
  acc[1][1] = __builtin_amdgcn_wmma_f32_16x16x32_bf16(
      false, f.a1, false, f.b1, (short)0, acc[1][1], false, false);
  acc[2][0] = __builtin_amdgcn_wmma_f32_16x16x32_bf16(
      false, f.a2, false, f.b0, (short)0, acc[2][0], false, false);
  acc[2][1] = __builtin_amdgcn_wmma_f32_16x16x32_bf16(
      false, f.a2, false, f.b1, (short)0, acc[2][1], false, false);
  acc[3][0] = __builtin_amdgcn_wmma_f32_16x16x32_bf16(
      false, f.a3, false, f.b0, (short)0, acc[3][0], false, false);
  acc[3][1] = __builtin_amdgcn_wmma_f32_16x16x32_bf16(
      false, f.a3, false, f.b1, (short)0, acc[3][1], false, false);
}

// ---------------------------------------------------------------------------
// Kernel 1: fp32 -> bf16 conversion (used for both x and W_enc)
// ---------------------------------------------------------------------------
__global__ __launch_bounds__(256) void k_conv_bf16(const float* __restrict__ src,
                                                   __bf16* __restrict__ dst,
                                                   long n) {
  long i = ((long)blockIdx.x * 256 + threadIdx.x) * 8;
  if (i + 8 <= n) {
    f4 a = *(const f4*)(src + i);
    f4 b = *(const f4*)(src + i + 4);
    bf16x8 o;
    o[0]=(__bf16)a[0]; o[1]=(__bf16)a[1]; o[2]=(__bf16)a[2]; o[3]=(__bf16)a[3];
    o[4]=(__bf16)b[0]; o[5]=(__bf16)b[1]; o[6]=(__bf16)b[2]; o[7]=(__bf16)b[3];
    *(bf16x8*)(dst + i) = o;
  }
}

// ---------------------------------------------------------------------------
// Kernel 2: W_dec (DOUT x HID) -> W_dec^T bf16 (HID x DOUT), LDS 32x32 tiles.
// ---------------------------------------------------------------------------
__global__ __launch_bounds__(256) void k_transpose_wdec(const float* __restrict__ wdec,
                                                        __bf16* __restrict__ wdt) {
  __shared__ float t[32][33];
  const int o0 = blockIdx.x * 32;
  const int h0 = blockIdx.y * 32;
  const int lj = threadIdx.x & 31;
  const int li = threadIdx.x >> 5;   // 0..7
  #pragma unroll
  for (int k = 0; k < 4; ++k) {
    int o = o0 + li + k * 8;
    t[li + k * 8][lj] = wdec[(size_t)o * HID + (h0 + lj)];
  }
  __syncthreads();
  #pragma unroll
  for (int k = 0; k < 4; ++k) {
    int h = h0 + li + k * 8;
    wdt[(size_t)h * DOUT + (o0 + lj)] = (__bf16)t[lj][li + k * 8];
  }
}

// ---------------------------------------------------------------------------
// Kernel 3: fused encoder GEMM (bf16 WMMA) + bias + JumpReLU + per-chunk top-64
// Grid: (ROWS/M_TILE, CHUNKS), 256 threads (8 waves).
// LDS: A-tile 64 x A_STRIDE bf16 (257KB, bank-conflict-free) + acts 32KB = 289KB.
// Software-pipelined K loop: fragments for step kk+32 are loaded before the
// 8 WMMAs of step kk issue, so LDS/global loads overlap the XDL pipeline.
// ---------------------------------------------------------------------------
__global__ __launch_bounds__(256) void k_encoder_topk(
    const __bf16* __restrict__ xb, const __bf16* __restrict__ wb,
    const float* __restrict__ b_enc, const float* __restrict__ thrv,
    unsigned* __restrict__ cand) {
  extern __shared__ char smem[];
  __bf16* Atile = (__bf16*)smem;                                    // [M_TILE][A_STRIDE]
  __bf16* acts  = (__bf16*)(smem + (size_t)M_TILE * A_STRIDE * 2);  // [M_TILE][H_CHUNK]

  const int tid  = threadIdx.x;
  const int wave = tid >> 5;
  const int lane = tid & 31;
  const int l16  = lane & 15;
  const int hi16 = lane >> 4;        // 0 or 1
  const int m0   = blockIdx.x * M_TILE;
  const int c0   = blockIdx.y * H_CHUNK;

  // ---- Stage A tile: bf16 x rows m0..m0+63 -> padded LDS rows ----
  {
    const __bf16* xrow = xb + (size_t)m0 * DIN;
    for (int i = tid; i < M_TILE * (DIN / 8); i += 256) {
      const int row = i >> 8;            // DIN/8 = 256 chunks of 8 per row
      const int c8  = i & 255;
      bf16x8 v = *(const bf16x8*)(xrow + (size_t)row * DIN + c8 * 8);
      *(bf16x8*)(Atile + (size_t)row * A_STRIDE + c8 * 8) = v;
    }
  }
  __syncthreads();

  const int koff = hi16 << 3;   // K sub-offset per lane half (ISA A/B layout)

  const __bf16* ag0 = Atile + (size_t)( 0 + l16) * A_STRIDE + koff;
  const __bf16* ag1 = Atile + (size_t)(16 + l16) * A_STRIDE + koff;
  const __bf16* ag2 = Atile + (size_t)(32 + l16) * A_STRIDE + koff;
  const __bf16* ag3 = Atile + (size_t)(48 + l16) * A_STRIDE + koff;

  // ---- GEMM over this chunk's 256 feature columns (single pass) ----
  const int n0 = c0 + wave * 32;
  v8f acc[4][2] = {};
  const __bf16* w0 = wb + (size_t)(n0 + l16) * DIN + koff; // B tile 0 (cols n0..n0+15)
  const __bf16* w1 = w0 + (size_t)16 * DIN;                // B tile 1 (cols n0+16..n0+31)

  Frags cur = load_frags(ag0, ag1, ag2, ag3, w0, w1, 0);
  #pragma unroll 2
  for (int kk = 0; kk < DIN - 32; kk += 32) {
    __builtin_prefetch(w0 + kk + 544, 0, 1);
    __builtin_prefetch(w1 + kk + 544, 0, 1);
    Frags nxt = load_frags(ag0, ag1, ag2, ag3, w0, w1, kk + 32);
    do_wmma(cur, acc);
    cur = nxt;
  }
  do_wmma(cur, acc);

  // ---- Epilogue: bias + JumpReLU, stash bf16 activations in LDS ----
  #pragma unroll
  for (int tt = 0; tt < 2; ++tt) {
    const int n = n0 + tt * 16 + l16;
    const float bias = b_enc[n];
    const float th   = thrv[n];
    const int col    = n - c0;
    #pragma unroll
    for (int g = 0; g < 4; ++g) {
      #pragma unroll
      for (int r = 0; r < 8; ++r) {
        const int row = g * 16 + r + (hi16 << 3);
        float v = acc[g][tt][r] + bias;
        v = (v > th) ? v : 0.0f;
        acts[(size_t)row * H_CHUNK + col] = (__bf16)v;
      }
    }
  }
  __syncthreads();

  // ---- Exact per-row top-64 of this 256-wide chunk (binary-search count) ----
  for (int rr = 0; rr < M_TILE / 8; ++rr) {
    const int row = wave * (M_TILE / 8) + rr;
    float vals[H_CHUNK / 32];
    #pragma unroll
    for (int j = 0; j < H_CHUNK / 32; ++j)
      vals[j] = (float)acts[(size_t)row * H_CHUNK + j * 32 + lane];

    float mx = 0.0f; int cpos = 0;
    #pragma unroll
    for (int j = 0; j < H_CHUNK / 32; ++j) {
      mx = fmaxf(mx, vals[j]);
      cpos += (vals[j] > 0.0f) ? 1 : 0;
    }
    mx   = wred_max(mx);
    cpos = wred_add(cpos);

    float sel_t = 0.0f;
    if (cpos > TOPK) {
      float lo = 0.0f, hi = mx;
      for (int it = 0; it < 20; ++it) {
        const float mid = 0.5f * (lo + hi);
        int c = 0;
        #pragma unroll
        for (int j = 0; j < H_CHUNK / 32; ++j) c += (vals[j] > mid) ? 1 : 0;
        c = wred_add(c);
        if (c > TOPK) lo = mid; else hi = mid;
      }
      sel_t = hi;   // count(> hi) <= 64
    }

    unsigned* crow = cand + (size_t)(m0 + row) * NCAND + (size_t)blockIdx.y * TOPK;
    crow[lane]      = 0u;   // pre-fill (padding when < 64 positives)
    crow[lane + 32] = 0u;
    int base = 0;
    for (int j = 0; j < H_CHUNK / 32 && base < TOPK; ++j) {
      const bool p = vals[j] > sel_t;
      const unsigned mask = (unsigned)__ballot(p);
      const int pos = base + __popc(mask & ((1u << lane) - 1u));
      if (p && pos < TOPK) {
        const unsigned gidx = (unsigned)(c0 + j * 32 + lane);  // < 16384, fits u16
        crow[pos] = ((unsigned)f2bf_bits(vals[j]) << 16) | gidx;
      }
      base += __popc(mask);
    }
  }
}

// ---------------------------------------------------------------------------
// Kernel 4: merge 4096 candidates -> final top-64, then sparse decoder + b_out.
// ---------------------------------------------------------------------------
__global__ __launch_bounds__(256) void k_merge_decode(
    const unsigned* __restrict__ cand, const __bf16* __restrict__ wdt,
    const float* __restrict__ b_out, float* __restrict__ out) {
  __shared__ unsigned scand[NCAND];
  __shared__ float    selv[TOPK];
  __shared__ unsigned seli[TOPK];

  const int tid  = threadIdx.x;
  const int wave = tid >> 5;
  const int lane = tid & 31;
  const long row = blockIdx.x;

  for (int i = tid; i < NCAND; i += 256)
    scand[i] = cand[(size_t)row * NCAND + i];
  __syncthreads();

  if (wave == 0) {
    float vals[NCAND / 32];
    #pragma unroll 8
    for (int j = 0; j < NCAND / 32; ++j)
      vals[j] = bf_bits2f((unsigned short)(scand[j * 32 + lane] >> 16));

    float mx = 0.0f; int cpos = 0;
    for (int j = 0; j < NCAND / 32; ++j) {
      mx = fmaxf(mx, vals[j]);
      cpos += (vals[j] > 0.0f) ? 1 : 0;
    }
    mx   = wred_max(mx);
    cpos = wred_add(cpos);

    float sel_t = 0.0f;
    if (cpos > TOPK) {
      float lo = 0.0f, hi = mx;
      for (int it = 0; it < 20; ++it) {
        const float mid = 0.5f * (lo + hi);
        int c = 0;
        for (int j = 0; j < NCAND / 32; ++j) c += (vals[j] > mid) ? 1 : 0;
        c = wred_add(c);
        if (c > TOPK) lo = mid; else hi = mid;
      }
      sel_t = hi;
    }

    selv[lane] = 0.0f; selv[lane + 32] = 0.0f;
    seli[lane] = 0u;   seli[lane + 32] = 0u;
    int base = 0;
    for (int j = 0; j < NCAND / 32 && base < TOPK; ++j) {
      const bool p = vals[j] > sel_t;
      const unsigned mask = (unsigned)__ballot(p);
      const int pos = base + __popc(mask & ((1u << lane) - 1u));
      if (p && pos < TOPK) {
        selv[pos] = vals[j];
        seli[pos] = scand[j * 32 + lane] & 0xFFFFu;
      }
      base += __popc(mask);
    }
  }
  __syncthreads();

  // Sparse decoder: out[row, o0:o0+8] = b_out + sum_k selv[k] * Wdec^T[seli[k], :]
  const int o0 = tid * 8;
  float acc[8];
  #pragma unroll
  for (int j = 0; j < 8; ++j) acc[j] = b_out[o0 + j];

  for (int k = 0; k < TOPK; ++k) {
    const float v = selv[k];
    const unsigned id = seli[k];
    const bf16x8 w = *(const bf16x8*)(wdt + (size_t)id * DOUT + o0);
    #pragma unroll
    for (int j = 0; j < 8; ++j) acc[j] += v * (float)w[j];
  }

  f4 s0, s1;
  s0[0]=acc[0]; s0[1]=acc[1]; s0[2]=acc[2]; s0[3]=acc[3];
  s1[0]=acc[4]; s1[1]=acc[5]; s1[2]=acc[6]; s1[3]=acc[7];
  float* orow = out + (size_t)row * DOUT + o0;
  *(f4*)orow       = s0;
  *(f4*)(orow + 4) = s1;
}

// ---------------------------------------------------------------------------
extern "C" void kernel_launch(void* const* d_in, const int* in_sizes, int n_in,
                              void* d_out, int out_size, void* d_ws, size_t ws_size,
                              hipStream_t stream) {
  const float* x     = (const float*)d_in[0];
  const float* W_enc = (const float*)d_in[1];
  const float* b_enc = (const float*)d_in[2];
  const float* thr   = (const float*)d_in[3];
  const float* W_dec = (const float*)d_in[4];
  const float* b_out = (const float*)d_in[5];
  float* out = (float*)d_out;

  // Workspace layout: xb 32MB | wb 64MB | wdt 64MB | cand 128MB  (288MB total)
  char* ws = (char*)d_ws;
  __bf16*  xbp  = (__bf16*)ws;
  __bf16*  wb   = (__bf16*)(ws + (size_t)ROWS * DIN * 2);
  __bf16*  wdt  = (__bf16*)(ws + (size_t)ROWS * DIN * 2 + (size_t)HID * DIN * 2);
  unsigned* cnd = (unsigned*)(ws + (size_t)ROWS * DIN * 2 + (size_t)HID * DIN * 4);

  // 1) x and W_enc fp32 -> bf16
  const long x_n    = (long)ROWS * DIN;                // 16,777,216
  const long wenc_n = (long)HID * DIN;                 // 33,554,432
  k_conv_bf16<<<(int)(x_n    / 8 / 256), 256, 0, stream>>>(x, xbp, x_n);
  k_conv_bf16<<<(int)(wenc_n / 8 / 256), 256, 0, stream>>>(W_enc, wb, wenc_n);

  // 2) W_dec -> W_dec^T bf16
  k_transpose_wdec<<<dim3(DOUT / 32, HID / 32), 256, 0, stream>>>(W_dec, wdt);

  // 3) Fused encoder GEMM (WMMA bf16) + JumpReLU + per-chunk top-64
  const size_t lds = (size_t)M_TILE * A_STRIDE * 2 + (size_t)M_TILE * H_CHUNK * 2; // ~289KB
  k_encoder_topk<<<dim3(ROWS / M_TILE, CHUNKS), 256, lds, stream>>>(xbp, wb, b_enc, thr, cnd);

  // 4) Merge candidates -> top-64, sparse decode + b_out
  k_merge_decode<<<ROWS, 256, 0, stream>>>(cnd, wdt, b_out, out);
}
